// SingleHeadCrossAttention_87411174408426
// MI455X (gfx1250) — compile-verified, matrix operand order checked by reading, not run on hardware
//
#include <hip/hip_runtime.h>
#include <hip/hip_bf16.h>
#include <math.h>
#include <stdint.h>

// ---------------------------------------------------------------------------
// SingleHeadCrossAttention for MI455X (gfx1250, wave32, WMMA).
// B=512, N=262144, D=256, D1=64, H=64, K=16.
// Big cost: Q @ keys^T (68.7 GFLOP, 256MB of keys) -> bf16 WMMA 16x16x32.
// Q projection kept in fp32 via v_wmma_f32_16x16x4_f32.
// finalize stages the scattered V_top gather through ASYNCcnt-tracked
// global_load_async_to_lds_b128 (CDNA5 async path).
// ---------------------------------------------------------------------------

#define BQ 512      // batch (queries)
#define NK 262144   // number of keys
#define DD 256      // feature dim
#define D1 64       // param feature dim
#define HH 64       // adapter hidden
#define TK 16       // top-k

#define NSLAB 512   // key slabs
#define SLAB  512   // keys per slab  (NSLAB*SLAB == NK)
#define CK    64    // keys per chunk staged in LDS
#define QG    256   // queries per scan workgroup
#define NCHUNK (SLAB / CK)

typedef float    f32x2  __attribute__((ext_vector_type(2)));
typedef float    f32x8  __attribute__((ext_vector_type(8)));
typedef unsigned u32x8  __attribute__((ext_vector_type(8)));
typedef __bf16   bf16x16 __attribute__((ext_vector_type(16)));

#define NEG_INF (-3.0e38f)

// round-to-nearest-even f32 -> bf16, packed pair (lo = first arg)
static __device__ __forceinline__ unsigned pk_bf16(float x, float y) {
    unsigned a = __builtin_bit_cast(unsigned, x);
    unsigned b = __builtin_bit_cast(unsigned, y);
    a = (a + 0x7FFFu + ((a >> 16) & 1u)) >> 16;
    b = (b + 0x7FFFu + ((b >> 16) & 1u)) >> 16;
    return (b << 16) | (a & 0xFFFFu);
}

// Pack 8 consecutive f32 (lo half K) + 8 consecutive f32 (hi half K) into the
// 16-bf16 operand register block (ISA 7.12.2 16-bit A/B layout per lane).
static __device__ __forceinline__ bf16x16 pack_tile(const float4 f0, const float4 f1,
                                                    const float4 f2, const float4 f3) {
    u32x8 u;
    u[0] = pk_bf16(f0.x, f0.y); u[1] = pk_bf16(f0.z, f0.w);
    u[2] = pk_bf16(f1.x, f1.y); u[3] = pk_bf16(f1.z, f1.w);
    u[4] = pk_bf16(f2.x, f2.y); u[5] = pk_bf16(f2.z, f2.w);
    u[6] = pk_bf16(f3.x, f3.y); u[7] = pk_bf16(f3.z, f3.w);
    return __builtin_bit_cast(bf16x16, u);
}

// ---------------------------------------------------------------------------
// Kernel 1: Q = query @ Wq^T  (fp32 WMMA 16x16x4). One wave per 16x16 tile.
// ---------------------------------------------------------------------------
__global__ __launch_bounds__(128) void qproj_kernel(const float* __restrict__ query,
                                                    const float* __restrict__ Wq,
                                                    float* __restrict__ Qout) {
    const int wid  = (blockIdx.x * blockDim.x + threadIdx.x) >> 5; // 0..511
    const int lane = threadIdx.x & 31;
    const int tm = wid >> 4;   // 32 row tiles
    const int tn = wid & 15;   // 16 col tiles
    const int m    = lane & 15;
    const int half = lane >> 4;

    f32x8 c = {0.f, 0.f, 0.f, 0.f, 0.f, 0.f, 0.f, 0.f};
    const float* qrow = query + (size_t)(tm * 16 + m) * DD;
    const float* wrow = Wq    + (size_t)(tn * 16 + m) * DD; // B[k][n] = Wq[n][k]

#pragma unroll 4
    for (int k0 = 0; k0 < DD; k0 += 4) {
        const int ka = k0 + 2 * half;   // A lanes 0-15: K{0,1}; 16-31: K{2,3}
        f32x2 av; av[0] = qrow[ka]; av[1] = qrow[ka + 1];
        f32x2 bv; bv[0] = wrow[ka]; bv[1] = wrow[ka + 1];
        c = __builtin_amdgcn_wmma_f32_16x16x4_f32(false, av, false, bv,
                                                  (short)0, c, false, false);
    }
#pragma unroll
    for (int v = 0; v < 8; ++v) {
        const int q = tm * 16 + v + 8 * half;
        Qout[(size_t)q * DD + tn * 16 + m] = c[v];
    }
}

// ---------------------------------------------------------------------------
// Kernel 2: score scan + per-slab top-16, software-pipelined key staging.
// Grid: NSLAB*2 WGs of 256 threads (8 waves). blockIdx = slab*2 + qgroup.
// ---------------------------------------------------------------------------
__global__ __launch_bounds__(256) void scan_kernel(const float* __restrict__ Q,
                                                   const float* __restrict__ keys,
                                                   float* __restrict__ cand_s,
                                                   int*   __restrict__ cand_i) {
    extern __shared__ char smem[];
    unsigned short (*kbuf)[264] = (unsigned short (*)[264])smem;          // 64*528B
    float (*sc)[CK + 1] = (float (*)[CK + 1])(smem + CK * 264 * 2);       // 256*65*4B

    const int slab = blockIdx.x >> 1;
    const int qg   = blockIdx.x & 1;
    const int tid  = threadIdx.x;
    const int lane = tid & 31;
    const int wave = tid >> 5;          // 0..7
    const int mrow = lane & 15;
    const int kh   = (lane >> 4) * 8;   // K sub-block per half-wave

    // ---- load this wave's two Q A-tiles (bf16), kept in registers -------
    bf16x16 a[2][8];
#pragma unroll
    for (int qt = 0; qt < 2; ++qt) {
        const int q = qg * QG + (wave * 2 + qt) * 16 + mrow;
        const float* qr = Q + (size_t)q * DD;
#pragma unroll
        for (int ks = 0; ks < 8; ++ks) {
            const int d0 = ks * 32 + kh;
            float4 f0 = *(const float4*)(qr + d0);
            float4 f1 = *(const float4*)(qr + d0 + 4);
            float4 f2 = *(const float4*)(qr + d0 + 16);
            float4 f3 = *(const float4*)(qr + d0 + 20);
            a[qt][ks] = pack_tile(f0, f1, f2, f3);
        }
    }

    // ---- running top-16 for query (qg*256 + tid) ------------------------
    float ts[TK]; int ti[TK];
#pragma unroll
    for (int r = 0; r < TK; ++r) { ts[r] = NEG_INF; ti[r] = 0; }
    float tmin = NEG_INF; int tpos = 0;

    const int ldr = tid >> 3;           // key-row this thread stages
    const int ldc = (tid & 7) * 32;     // 32 dims per thread

    // preload chunk 0 into registers (software pipeline stage)
    float4 g[8];
    {
        const float* kr = keys + (size_t)(slab * SLAB + ldr) * DD + ldc;
#pragma unroll
        for (int u = 0; u < 8; ++u) g[u] = *(const float4*)(kr + u * 4);
    }

    for (int chunk = 0; chunk < NCHUNK; ++chunk) {
        const int kc0 = slab * SLAB + chunk * CK;

        // convert pipelined registers -> bf16 LDS
#pragma unroll
        for (int u = 0; u < 8; u += 2) {
            uint4 p;
            p.x = pk_bf16(g[u].x, g[u].y);     p.y = pk_bf16(g[u].z, g[u].w);
            p.z = pk_bf16(g[u + 1].x, g[u + 1].y); p.w = pk_bf16(g[u + 1].z, g[u + 1].w);
            *(uint4*)&kbuf[ldr][ldc + u * 4] = p;
        }
        // issue next chunk's global loads now; the loadcnt wait lands at the
        // next iteration's convert, hiding latency under the WMMA block.
        if (chunk + 1 < NCHUNK) {
            const float* kn = keys + (size_t)(kc0 + CK + ldr) * DD + ldc;
#pragma unroll
            for (int u = 0; u < 8; ++u) g[u] = *(const float4*)(kn + u * 4);
        }
        if (chunk + 2 < NCHUNK)   // warm L2 one more chunk ahead
            __builtin_prefetch(keys + (size_t)(kc0 + 2 * CK + ldr) * DD + ldc, 0, 0);
        __syncthreads();

        // scores: 16 query tiles x 4 key tiles, K accumulated over 8 WMMAs
#pragma unroll
        for (int kt = 0; kt < CK / 16; ++kt) {
            bf16x16 btile[8];
            const int krow = kt * 16 + mrow;
#pragma unroll
            for (int ks = 0; ks < 8; ++ks) {
                const int d0 = ks * 32 + kh;
                uint4 lo = *(const uint4*)&kbuf[krow][d0];
                uint4 hi = *(const uint4*)&kbuf[krow][d0 + 16];
                u32x8 u; u[0]=lo.x; u[1]=lo.y; u[2]=lo.z; u[3]=lo.w;
                         u[4]=hi.x; u[5]=hi.y; u[6]=hi.z; u[7]=hi.w;
                btile[ks] = __builtin_bit_cast(bf16x16, u);
            }
#pragma unroll
            for (int qt = 0; qt < 2; ++qt) {
                f32x8 c = {0.f, 0.f, 0.f, 0.f, 0.f, 0.f, 0.f, 0.f};
#pragma unroll
                for (int ks = 0; ks < 8; ++ks)
                    c = __builtin_amdgcn_wmma_f32_16x16x32_bf16(
                            false, a[qt][ks], false, btile[ks],
                            (short)0, c, false, false);
                const int qbase = (wave * 2 + qt) * 16 + 8 * (lane >> 4);
                const int ncol  = kt * 16 + mrow;
#pragma unroll
                for (int v = 0; v < 8; ++v) sc[qbase + v][ncol] = c[v];
            }
        }
        __syncthreads();

        // threshold-gated top-16 update (thread t == local query t)
        for (int j = 0; j < CK; ++j) {
            const float s = sc[tid][j];
            if (s > tmin) {
                ts[tpos] = s; ti[tpos] = kc0 + j;
                tmin = ts[0]; tpos = 0;
#pragma unroll
                for (int r = 1; r < TK; ++r)
                    if (ts[r] < tmin) { tmin = ts[r]; tpos = r; }
            }
        }
        __syncthreads();
    }

    const size_t base = ((size_t)slab * BQ + (qg * QG + tid)) * TK;
#pragma unroll
    for (int r = 0; r < TK; ++r) { cand_s[base + r] = ts[r]; cand_i[base + r] = ti[r]; }
}

// ---------------------------------------------------------------------------
// Kernel 3: merge per-slab candidates -> global top-16 indices per query.
// ---------------------------------------------------------------------------
__global__ __launch_bounds__(256) void merge_kernel(const float* __restrict__ cand_s,
                                                    const int*   __restrict__ cand_i,
                                                    int* __restrict__ topk_idx) {
    const int b = blockIdx.x * 256 + threadIdx.x;
    if (b >= BQ) return;
    float ts[TK]; int ti[TK];
#pragma unroll
    for (int r = 0; r < TK; ++r) { ts[r] = NEG_INF; ti[r] = 0; }
    float tmin = NEG_INF; int tpos = 0;
    for (int s = 0; s < NSLAB; ++s) {
        const size_t base = ((size_t)s * BQ + b) * TK;
        for (int j = 0; j < TK; ++j) {
            const float v = cand_s[base + j];
            if (v > tmin) {
                ts[tpos] = v; ti[tpos] = cand_i[base + j];
                tmin = ts[0]; tpos = 0;
#pragma unroll
                for (int r = 1; r < TK; ++r)
                    if (ts[r] < tmin) { tmin = ts[r]; tpos = r; }
            }
        }
    }
#pragma unroll
    for (int r = 0; r < TK; ++r) topk_idx[b * TK + r] = ti[r];
}

// ---------------------------------------------------------------------------
// Kernel 4: gather + adapter MLP + LayerNorm + softmax(16) + output.
// One 256-thread WG per query; thread t owns output dim d = t.
// V_top gather goes straight to LDS via async loads (ASYNCcnt).
// ---------------------------------------------------------------------------
__global__ __launch_bounds__(256) void finalize_kernel(
        const float* __restrict__ Q, const float* __restrict__ keys,
        const float* __restrict__ values, const float* __restrict__ pf,
        const float* __restrict__ w1, const float* __restrict__ b1,
        const float* __restrict__ w2, const float* __restrict__ b2,
        const float* __restrict__ gamma, const float* __restrict__ beta,
        const int* __restrict__ topk_idx, float* __restrict__ out) {
    __shared__ float vtop[TK][DD + 4];     // row stride 1040B: 16B aligned
    __shared__ float pfb[D1];
    __shared__ float hbuf[TK][HH + 1];
    __shared__ float sraw[TK];
    __shared__ float redA[256], redB[256];
    __shared__ float mu[TK], rsig[TK];

    const int b = blockIdx.x;
    const int t = threadIdx.x;
    const int lane = t & 31, wave = t >> 5;

    int idx[TK];
#pragma unroll
    for (int k = 0; k < TK; ++k) idx[k] = topk_idx[b * TK + k];

    // stage V_top rows: 1024 async b128 copies, 4 per thread, memory -> LDS
    // without touching VGPRs (tracked by ASYNCcnt).
#pragma unroll
    for (int it = 0; it < 4; ++it) {
        const int e = it * 1024 + t * 4;        // float index, 16B granules
        const int k = e >> 8, d = e & (DD - 1);
        const unsigned ldsoff = (unsigned)(uintptr_t)&vtop[k][d];
        const float* gp = values + (size_t)idx[k] * DD + d;
        asm volatile("global_load_async_to_lds_b128 %0, %1, off"
                     :: "v"(ldsoff), "v"(gp) : "memory");
    }
    if (t < D1) pfb[t] = pf[b * D1 + t];

    // fp32 attention logits: wave w -> k = 2w, 2w+1 (shuffle reduce)
    const float* qr = Q + (size_t)b * DD;
#pragma unroll
    for (int r = 0; r < 2; ++r) {
        const int k = wave * 2 + r;
        const float* krow = keys + (size_t)idx[k] * DD;
        float acc = 0.f;
        for (int u = lane; u < DD; u += 32) acc += qr[u] * krow[u];
#pragma unroll
        for (int s = 16; s > 0; s >>= 1) acc += __shfl_xor(acc, s, 32);
        if (lane == 0) sraw[k] = acc * (1.0f / 16.0f);   // /sqrt(256)
    }
    asm volatile("s_wait_asynccnt 0x0" ::: "memory");    // V_top resident in LDS
    __syncthreads();

    // adapter layer 1: h = relu([V_top, pf] @ w1^T + b1); 1024 dots, 4/thread
#pragma unroll
    for (int ee = 0; ee < 4; ++ee) {
        const int e = t * 4 + ee, k = e >> 6, j = e & 63;
        const float* w1r = w1 + (size_t)j * (DD + D1);
        float acc = b1[j];
        for (int f = 0; f < DD; ++f) acc += vtop[k][f] * w1r[f];
#pragma unroll 8
        for (int f = 0; f < D1; ++f) acc += pfb[f] * w1r[DD + f];
        hbuf[k][j] = fmaxf(acc, 0.f);
    }
    __syncthreads();

    // adapter layer 2 for dim d=t, all 16 k at once
    const int d = t;
    float aacc[TK];
    const float bb2 = b2[d];
#pragma unroll
    for (int k = 0; k < TK; ++k) aacc[k] = bb2;
    for (int j = 0; j < HH; ++j) {
        const float wv = w2[(size_t)d * HH + j];
#pragma unroll
        for (int k = 0; k < TK; ++k) aacc[k] += hbuf[k][j] * wv;
    }

    // LayerNorm stats per k (tree reduce over 256 dims)
    for (int k = 0; k < TK; ++k) {
        redA[t] = aacc[k]; redB[t] = aacc[k] * aacc[k];
        __syncthreads();
        for (int s = 128; s > 0; s >>= 1) {
            if (t < s) { redA[t] += redA[t + s]; redB[t] += redB[t + s]; }
            __syncthreads();
        }
        if (t == 0) {
            const float m = redA[0] * (1.0f / DD);
            const float v = redB[0] * (1.0f / DD) - m * m;
            mu[k] = m; rsig[k] = rsqrtf(v + 1e-5f);
        }
        __syncthreads();
    }

    // softmax over 16 (redundant per thread) + weighted sum
    float smax = NEG_INF;
#pragma unroll
    for (int k = 0; k < TK; ++k) smax = fmaxf(smax, sraw[k]);
    float sexp[TK], ssum = 0.f;
#pragma unroll
    for (int k = 0; k < TK; ++k) { sexp[k] = expf(sraw[k] - smax); ssum += sexp[k]; }
    const float inv = 1.0f / ssum;
    const float gd = gamma[d], bd = beta[d];
    float o = 0.f;
#pragma unroll
    for (int k = 0; k < TK; ++k) {
        const float ln = (aacc[k] - mu[k]) * rsig[k] * gd + bd;
        o += sexp[k] * inv * (vtop[k][d] + ln);
    }
    out[(size_t)b * DD + d] = o;
}

// ---------------------------------------------------------------------------
extern "C" void kernel_launch(void* const* d_in, const int* in_sizes, int n_in,
                              void* d_out, int out_size, void* d_ws, size_t ws_size,
                              hipStream_t stream) {
    const float* query = (const float*)d_in[0];
    const float* keys  = (const float*)d_in[1];
    const float* values= (const float*)d_in[2];
    const float* pf    = (const float*)d_in[3];
    const float* Wq    = (const float*)d_in[4];
    const float* w1    = (const float*)d_in[5];
    const float* b1    = (const float*)d_in[6];
    const float* w2    = (const float*)d_in[7];
    const float* b2    = (const float*)d_in[8];
    const float* gam   = (const float*)d_in[9];
    const float* bet   = (const float*)d_in[10];
    float* out = (float*)d_out;

    char* ws = (char*)d_ws;
    float* Qbuf   = (float*)ws;                               // 512*256*4   = 512 KB
    float* cand_s = (float*)(ws + 524288);                    // 512*512*16*4 = 16 MB
    int*   cand_i = (int*)  (ws + 524288 + 16777216);         // 16 MB
    int*   topk   = (int*)  (ws + 524288 + 2 * 16777216);     // 512*16*4

    qproj_kernel<<<128, 128, 0, stream>>>(query, Wq, Qbuf);

    const size_t scan_lds = (size_t)CK * 264 * 2 + (size_t)QG * (CK + 1) * 4; // 100,352 B
    scan_kernel<<<NSLAB * 2, 256, scan_lds, stream>>>(Qbuf, keys, cand_s, cand_i);

    merge_kernel<<<2, 256, 0, stream>>>(cand_s, cand_i, topk);

    finalize_kernel<<<BQ, 256, 0, stream>>>(Qbuf, keys, values, pf, w1, b1, w2, b2,
                                            gam, bet, topk, out);
}
